// REGCN_29231547417246
// MI455X (gfx1250) — matrix operand order repaired and checked by reading, not compiled, not run on hardware
//
#include <hip/hip_runtime.h>
#include <cmath>

// ---------------- problem constants ----------------
constexpr int NN  = 30000;   // nodes
constexpr int EE  = 100000;  // edges per relation
constexpr int ELL = 20000;   // label edges per relation
constexpr int RR  = 8;       // relations
constexpr int KD  = 200;     // C = H1 = H2
constexpr int KP  = 224;     // K padded to multiple of 32 (7 k-tiles)
constexpr int NT  = 14;      // padded col-tiles (cols 0..223, 7 tile-pairs)
constexpr int OCC = 50;      // decoder channels
constexpr int BPER = NT * 7 * 32 * 16;  // halfs per fragment-packed B matrix (50176)
constexpr float SLOPE_F = 0.2291666666666667f;  // (1/8 + 1/3)/2

typedef __attribute__((ext_vector_type(16))) _Float16 v16h;
typedef __attribute__((ext_vector_type(8)))  _Float16 v8h;
typedef __attribute__((ext_vector_type(8)))  float    v8f;

// ---------------- small prep kernels ----------------

// per-relation in-degree counts (same edge set both layers)
__global__ void k_counts(const int* __restrict__ ei, float* __restrict__ cnt) {
  int t = blockIdx.x * blockDim.x + threadIdx.x;
  if (t >= RR * EE) return;
  int r = t / EE, e = t - r * EE;
  int dst = ei[(size_t)(r * 2 + 1) * EE + e];
  atomicAdd(&cnt[(size_t)r * NN + dst], 1.0f);
}

__global__ void k_invc(const float* __restrict__ cnt, float* __restrict__ invc) {
  int t = blockIdx.x * blockDim.x + threadIdx.x;
  if (t >= RR * NN) return;
  invc[t] = 1.0f / fmaxf(cnt[t], 1.0f);
}

// pack W (or sum_r W_r) into WMMA B-fragment order:
// Bf[m][ntile][kt][lane][j] = W[m or sum][col][k], col = ntile*16+(lane&15),
// k = kt*32 + (lane>>4)*8 + (j&7) + (j>>3)*16  (zero padded outside 200x200)
__global__ void k_pack_Bfrag(const float* __restrict__ W, _Float16* __restrict__ Bf,
                             int nmat, int sumR) {
  int t = blockIdx.x * blockDim.x + threadIdx.x;
  if (t >= nmat * BPER) return;
  int m = t / BPER;
  int rem = t - m * BPER;
  int j = rem & 15;  rem >>= 4;
  int lane = rem & 31;  rem >>= 5;
  int kt = rem % 7;
  int ntile = rem / 7;
  int col = ntile * 16 + (lane & 15);
  int k = kt * 32 + ((lane >> 4) << 3) + (j & 7) + ((j >> 3) << 4);
  float s = 0.0f;
  if (col < KD && k < KD) {
    if (sumR) {
      for (int r = 0; r < RR; ++r) s += W[((size_t)r * KD + col) * KD + k];
    } else {
      s = W[((size_t)m * KD + col) * KD + k];
    }
  }
  Bf[t] = (_Float16)s;
}

__global__ void k_bmean(const float* __restrict__ b, float* __restrict__ bm) {
  int h = threadIdx.x;
  if (h >= KD) return;
  float s = 0.0f;
  for (int r = 0; r < RR; ++r) s += b[r * KD + h];
  bm[h] = s * (1.0f / RR);
}

// gate helper: svec[c] = sum_h uW[h][c]; svec[200] = sum(ub)
__global__ void k_svec(const float* __restrict__ uW, const float* __restrict__ ub,
                       float* __restrict__ sv) {
  int t = threadIdx.x;
  if (t < KD) {
    float s = 0.0f;
    for (int h = 0; h < KD; ++h) s += uW[h * KD + t];
    sv[t] = s;
  } else if (t == KD) {
    float s = 0.0f;
    for (int h = 0; h < KD; ++h) s += ub[h];
    sv[KD] = s;
  }
}

// pw[k] = post_W[0][k] + post_W[1][k]; pw[150] = post_b[0]+post_b[1]
__global__ void k_pw(const float* __restrict__ pW, const float* __restrict__ pb,
                     float* __restrict__ pwb) {
  int t = threadIdx.x;
  if (t < 150)       pwb[t] = pW[t] + pW[150 + t];
  else if (t == 150) pwb[150] = pb[0] + pb[1];
}

// ---------------- aggregation scatter ----------------
// one thread handles 8 channels of one edge: magg[dst][c..] += Xin[src][c..]
__global__ void k_scatter(const float* __restrict__ Xin, const int* __restrict__ ei,
                          int r, float* __restrict__ magg) {
  int t = blockIdx.x * blockDim.x + threadIdx.x;
  if (t >= EE * 25) return;
  int e = t / 25, ch = t - e * 25;
  int src = ei[(size_t)(r * 2) * EE + e];
  int dst = ei[(size_t)(r * 2 + 1) * EE + e];
  const float* xs = Xin + (size_t)src * KD + ch * 8;
  float* md = magg + (size_t)dst * KD + ch * 8;
#pragma unroll
  for (int j = 0; j < 8; ++j) atomicAdd(&md[j], xs[j]);
}

// ---------------- A-operand conversion: f32 [N][200] -> f16 [N][224] padded ----
// rowscale folds the 1/max(cnt,1) mean divide into the conversion
__global__ void k_convA(const float* __restrict__ A, const float* __restrict__ rowscale,
                        _Float16* __restrict__ Ah) {
  int t = blockIdx.x * blockDim.x + threadIdx.x;
  if (t >= NN * (KP / 4)) return;        // 4 halfs per thread
  int n = t / (KP / 4), c4 = (t - n * (KP / 4)) * 4;
  float rs = rowscale ? rowscale[n] : 1.0f;
  const float* ar = A + (size_t)n * KD;
  _Float16* dr = Ah + (size_t)n * KP + c4;
#pragma unroll
  for (int j = 0; j < 4; ++j) {
    int k = c4 + j;
    dr[j] = (k < KD) ? (_Float16)(ar[k] * rs) : (_Float16)0.0f;
  }
}

// ---------------- WMMA GEMM: one wave computes a 16x32 output tile ----------------
// Out[16 rows][two 16-col tiles] += Ah @ Bf ; optional epilogue (/R + bias, RReLU)
__global__ __launch_bounds__(32)
void k_gemm_tile(const _Float16* __restrict__ Ah,   // [NN][KP] f16 padded (rowscaled)
                 const _Float16* __restrict__ Bf,   // fragment-packed [NT][7][32][16]
                 const float* __restrict__ bias,
                 float* __restrict__ Out, int finalize) {
  const int lane  = threadIdx.x;            // 0..31, wave32
  const int mtile = blockIdx.x;             // NN/16 = 1875
  const int npair = blockIdx.y;             // 0..6 -> col tiles 2*npair, 2*npair+1
  const int khalf = lane >> 4;
  const int lidx  = lane & 15;
  const int col0  = (npair << 5) + lidx;    // first tile column
  const int col1  = col0 + 16;

  // load running accumulation (C layout: VGPR i -> row mtile*16+i+khalf*8, col lane&15)
  v8f c0, c1;
#pragma unroll
  for (int i = 0; i < 8; ++i) {
    int orow = (mtile << 4) + i + (khalf << 3);
    c0[i] = (col0 < KD) ? Out[(size_t)orow * KD + col0] : 0.0f;
    c1[i] = (col1 < KD) ? Out[(size_t)orow * KD + col1] : 0.0f;
  }

  const _Float16* ap  = Ah + (size_t)((mtile << 4) + lidx) * KP + (khalf << 3);
  const _Float16* bp0 = Bf + ((size_t)(2 * npair) * 7 * 32 + lane) * 16;
  const _Float16* bp1 = bp0 + (size_t)7 * 32 * 16;

#pragma unroll
  for (int kt = 0; kt < 7; ++kt) {
    v8h alo = *(const v8h*)(ap + kt * 32);        // K = kt*32 + khalf*8 + 0..7
    v8h ahi = *(const v8h*)(ap + kt * 32 + 16);   // K = kt*32 + 16 + khalf*8 + 0..7
    v16h a = __builtin_shufflevector(alo, ahi, 0, 1, 2, 3, 4, 5, 6, 7,
                                     8, 9, 10, 11, 12, 13, 14, 15);
    v16h b0 = *(const v16h*)(bp0 + (size_t)kt * 32 * 16);
    v16h b1 = *(const v16h*)(bp1 + (size_t)kt * 32 * 16);
#if defined(__gfx1250__)
    c0 = __builtin_amdgcn_wmma_f32_16x16x32_f16(false, a, false, b0, (short)0, c0,
                                                false, false);
    c1 = __builtin_amdgcn_wmma_f32_16x16x32_f16(false, a, false, b1, (short)0, c1,
                                                false, false);
#endif
  }

#pragma unroll
  for (int i = 0; i < 8; ++i) {
    int orow = (mtile << 4) + i + (khalf << 3);
    if (col0 < KD) {
      float v = c0[i];
      if (finalize) {
        v = v * (1.0f / RR) + bias[col0];
        v = (v >= 0.0f) ? v : v * SLOPE_F;
      }
      Out[(size_t)orow * KD + col0] = v;
    }
    if (col1 < KD) {
      float v = c1[i];
      if (finalize) {
        v = v * (1.0f / RR) + bias[col1];
        v = (v >= 0.0f) ? v : v * SLOPE_F;
      }
      Out[(size_t)orow * KD + col1] = v;
    }
  }
}

// ---------------- temporal gate: u = sigmoid(past . svec + sb) ----------------
__global__ void k_gate(const float* __restrict__ outv, const float* __restrict__ past,
                       const float* __restrict__ sv, const int* __restrict__ snap,
                       float* __restrict__ cur) {
  int n = blockIdx.x * blockDim.x + threadIdx.x;
  if (n >= NN) return;
  const float* pr = past + (size_t)n * KD;
  const float* ov = outv + (size_t)n * KD;
  float* cr = cur + (size_t)n * KD;
  if (*snap == 0) {
    for (int h = 0; h < KD; ++h) cr[h] = ov[h];
    return;
  }
  float d = sv[KD];
  for (int h = 0; h < KD; ++h) d += pr[h] * sv[h];
  float u = 1.0f / (1.0f + expf(-d));
  for (int h = 0; h < KD; ++h) cr[h] = u * ov[h] + (1.0f - u) * pr[h];
}

// ---------------- relation pooling: avg[r][h] = mean_e cur2[dst[r][e]][h] -------
__global__ void k_avg(const float* __restrict__ cur2, const int* __restrict__ ei,
                      float* __restrict__ avg) {
  int h = blockIdx.x, r = blockIdx.y;
  const int* dst = ei + (size_t)(r * 2 + 1) * EE;
  __shared__ float red[256];
  float s = 0.0f;
  for (int e = threadIdx.x; e < EE; e += 256) s += cur2[(size_t)dst[e] * KD + h];
  red[threadIdx.x] = s;
  __syncthreads();
  for (int st = 128; st > 0; st >>= 1) {
    if (threadIdx.x < st) red[threadIdx.x] += red[threadIdx.x + st];
    __syncthreads();
  }
  if (threadIdx.x == 0) avg[r * KD + h] = red[0] / (float)EE;
}

// ---------------- GRU cell / linear over Rmat = [avg | rel_emb] ----------------
__global__ void k_gru(const float* __restrict__ avg, const float* __restrict__ rel,
                      const float* __restrict__ pastR,
                      const float* __restrict__ Wih, const float* __restrict__ Whh,
                      const float* __restrict__ bih, const float* __restrict__ bhh,
                      const float* __restrict__ lW, const float* __restrict__ lb,
                      const int* __restrict__ snap, float* __restrict__ curR) {
  int r = blockIdx.x;
  __shared__ float sm[600];   // [0:200)=avg, [200:400)=rel_emb, [400:600)=pastR
  for (int t = threadIdx.x; t < 600; t += blockDim.x)
    sm[t] = (t < 200) ? avg[r * KD + t]
          : (t < 400) ? rel[r * KD + (t - 200)]
                      : pastR[r * KD + (t - 400)];
  __syncthreads();
  int h = threadIdx.x;
  if (h >= KD) return;
  if (*snap == 0) {
    float s = lb[h];
    for (int k = 0; k < 400; ++k) s += sm[k] * lW[h * 400 + k];
    curR[r * KD + h] = s;
    return;
  }
  float gir = bih[h], giz = bih[200 + h], gin = bih[400 + h];
  for (int k = 0; k < 400; ++k) {
    float x = sm[k];
    gir += x * Wih[(h)       * 400 + k];
    giz += x * Wih[(200 + h) * 400 + k];
    gin += x * Wih[(400 + h) * 400 + k];
  }
  float ghr = bhh[h], ghz = bhh[200 + h], ghn = bhh[400 + h];
  for (int k = 0; k < 200; ++k) {
    float x = sm[400 + k];
    ghr += x * Whh[(h)       * 200 + k];
    ghz += x * Whh[(200 + h) * 200 + k];
    ghn += x * Whh[(400 + h) * 200 + k];
  }
  float rg = 1.0f / (1.0f + expf(-(gir + ghr)));
  float zg = 1.0f / (1.0f + expf(-(giz + ghz)));
  float ng = tanhf(gin + rg * ghn);
  curR[r * KD + h] = (1.0f - zg) * ng + zg * sm[400 + h];
}

// rterm[r] = sum_o (convr_W[o].curR[r] + convr_b[o]) * pw[50+o]
__global__ void k_rterm(const float* __restrict__ curR, const float* __restrict__ cW,
                        const float* __restrict__ cb, const float* __restrict__ pwb,
                        float* __restrict__ rterm) {
  int r = blockIdx.x, o = threadIdx.x;
  __shared__ float red[64];
  float v = 0.0f;
  if (o < OCC) {
    float s = cb[o];
    for (int k = 0; k < KD; ++k) s += cW[o * KD + k] * curR[r * KD + k];
    v = s * pwb[OCC + o];
  }
  red[o] = v;
  __syncthreads();
  for (int st = 32; st > 0; st >>= 1) {
    if (o < st) red[o] += red[o + st];
    __syncthreads();
  }
  if (o == 0) rterm[r] = red[0];
}

// decoder, faithful to the raw (H2,EL) reshape in the original torch code:
// hc[e][o] = Y[yrow][ycol], Y = convh_W @ Xm + b, Xm[i][j] = cur2[eli0[i*100+j/200]][j%200]
__global__ void k_decode(const float* __restrict__ cur2, const int* __restrict__ eli,
                         const float* __restrict__ chW, const float* __restrict__ chb,
                         const float* __restrict__ pwb, const float* __restrict__ rterm,
                         float* __restrict__ out) {
  int r = blockIdx.y;
  int e = blockIdx.x * blockDim.x + threadIdx.x;
  if (e >= ELL) return;
  const int* e0 = eli + (size_t)(r * 2) * ELL;
  const int* e1 = e0 + ELL;
  float acc = rterm[r] + pwb[150];
#pragma unroll 1
  for (int o = 0; o < OCC; ++o) {
    int idx  = e * OCC + o;
    int yrow = idx / ELL;
    int ycol = idx - yrow * ELL;
    int q    = ycol / KD;
    int ch   = ycol - q * KD;
    float sh = chb[yrow], st = chb[yrow];
    const float* wrow = chW + (size_t)yrow * KD;
    for (int i = 0; i < KD; ++i) {
      float w = wrow[i];
      int slot = i * 100 + q;
      sh += w * cur2[(size_t)e0[slot] * KD + ch];
      st += w * cur2[(size_t)e1[slot] * KD + ch];
    }
    acc += sh * pwb[o] + st * pwb[100 + o];
  }
  out[(size_t)r * ELL + e] = acc;
}

// ---------------- host side ----------------
extern "C" void kernel_launch(void* const* d_in, const int* in_sizes, int n_in,
                              void* d_out, int out_size, void* d_ws, size_t ws_size,
                              hipStream_t stream) {
  const float* x     = (const float*)d_in[0];
  const int*   ei    = (const int*)d_in[1];
  const int*   eli   = (const int*)d_in[2];
  const float* past1 = (const float*)d_in[3];
  const float* past2 = (const float*)d_in[4];
  const float* pastR = (const float*)d_in[5];
  const float* W1l   = (const float*)d_in[6];
  const float* b1l   = (const float*)d_in[7];
  const float* W1r   = (const float*)d_in[8];
  const float* W2l   = (const float*)d_in[9];
  const float* b2l   = (const float*)d_in[10];
  const float* W2r   = (const float*)d_in[11];
  const float* u1W   = (const float*)d_in[12];
  const float* u1b   = (const float*)d_in[13];
  const float* u2W   = (const float*)d_in[14];
  const float* u2b   = (const float*)d_in[15];
  const float* rel   = (const float*)d_in[16];
  const float* gWih  = (const float*)d_in[17];
  const float* gWhh  = (const float*)d_in[18];
  const float* gbih  = (const float*)d_in[19];
  const float* gbhh  = (const float*)d_in[20];
  const float* lW    = (const float*)d_in[21];
  const float* lb    = (const float*)d_in[22];
  const float* chW   = (const float*)d_in[23];
  const float* chb   = (const float*)d_in[24];
  const float* crW   = (const float*)d_in[25];
  const float* crb   = (const float*)d_in[26];
  const float* pW    = (const float*)d_in[27];
  const float* pb    = (const float*)d_in[28];
  const int*   snap  = (const int*)d_in[29];
  float* out = (float*)d_out;

  // workspace carve-up (~140 MB)
  char* w = (char*)d_ws;
  size_t off = 0;
  auto take = [&](size_t bytes) -> char* {
    char* p = w + off;
    off = (off + bytes + 255) & ~(size_t)255;
    return p;
  };
  float*    cnt   = (float*)take((size_t)RR * NN * 4);
  float*    invc  = (float*)take((size_t)RR * NN * 4);
  float*    magg  = (float*)take((size_t)NN * KD * 4);
  float*    out1  = (float*)take((size_t)NN * KD * 4);
  float*    cur1  = (float*)take((size_t)NN * KD * 4);
  float*    out2  = (float*)take((size_t)NN * KD * 4);
  float*    cur2  = (float*)take((size_t)NN * KD * 4);
  _Float16* Ah    = (_Float16*)take((size_t)NN * KP * 2);
  _Float16* Bl1   = (_Float16*)take((size_t)RR * BPER * 2);
  _Float16* Brs1  = (_Float16*)take((size_t)BPER * 2);
  _Float16* Bl2   = (_Float16*)take((size_t)RR * BPER * 2);
  _Float16* Brs2  = (_Float16*)take((size_t)BPER * 2);
  float*    bm1   = (float*)take(256 * 4);
  float*    bm2   = (float*)take(256 * 4);
  float*    sv1   = (float*)take(256 * 4);
  float*    sv2   = (float*)take(256 * 4);
  float*    avg   = (float*)take(RR * KD * 4);
  float*    curR  = (float*)take(RR * KD * 4);
  float*    rterm = (float*)take(RR * 4);
  float*    pwb   = (float*)take(256 * 4);
  (void)in_sizes; (void)n_in; (void)out_size; (void)ws_size;

  const dim3 gGrid(NN / 16, 7);   // 1875 row-tiles x 7 col-tile-pairs
  const int  convBlocks = (NN * (KP / 4) + 255) / 256;

  // prep
  hipMemsetAsync(cnt, 0, (size_t)RR * NN * 4, stream);
  k_counts<<<(RR * EE + 255) / 256, 256, 0, stream>>>(ei, cnt);
  k_invc<<<(RR * NN + 255) / 256, 256, 0, stream>>>(cnt, invc);
  k_pack_Bfrag<<<(RR * BPER + 255) / 256, 256, 0, stream>>>(W1l, Bl1, RR, 0);
  k_pack_Bfrag<<<(RR * BPER + 255) / 256, 256, 0, stream>>>(W2l, Bl2, RR, 0);
  k_pack_Bfrag<<<(BPER + 255) / 256, 256, 0, stream>>>(W1r, Brs1, 1, 1);
  k_pack_Bfrag<<<(BPER + 255) / 256, 256, 0, stream>>>(W2r, Brs2, 1, 1);
  k_bmean<<<1, 256, 0, stream>>>(b1l, bm1);
  k_bmean<<<1, 256, 0, stream>>>(b2l, bm2);
  k_svec<<<1, 256, 0, stream>>>(u1W, u1b, sv1);
  k_svec<<<1, 256, 0, stream>>>(u2W, u2b, sv2);
  k_pw<<<1, 256, 0, stream>>>(pW, pb, pwb);

  // ---- layer 1 ----
  hipMemsetAsync(out1, 0, (size_t)NN * KD * 4, stream);
  for (int r = 0; r < RR; ++r) {
    hipMemsetAsync(magg, 0, (size_t)NN * KD * 4, stream);
    k_scatter<<<(EE * 25 + 255) / 256, 256, 0, stream>>>(x, ei, r, magg);
    k_convA<<<convBlocks, 256, 0, stream>>>(magg, invc + (size_t)r * NN, Ah);
    k_gemm_tile<<<gGrid, 32, 0, stream>>>(Ah, Bl1 + (size_t)r * BPER, nullptr, out1, 0);
  }
  k_convA<<<convBlocks, 256, 0, stream>>>(x, nullptr, Ah);
  k_gemm_tile<<<gGrid, 32, 0, stream>>>(Ah, Brs1, bm1, out1, 1);
  k_gate<<<(NN + 255) / 256, 256, 0, stream>>>(out1, past1, sv1, snap, cur1);

  // ---- layer 2 ----
  hipMemsetAsync(out2, 0, (size_t)NN * KD * 4, stream);
  for (int r = 0; r < RR; ++r) {
    hipMemsetAsync(magg, 0, (size_t)NN * KD * 4, stream);
    k_scatter<<<(EE * 25 + 255) / 256, 256, 0, stream>>>(cur1, ei, r, magg);
    k_convA<<<convBlocks, 256, 0, stream>>>(magg, invc + (size_t)r * NN, Ah);
    k_gemm_tile<<<gGrid, 32, 0, stream>>>(Ah, Bl2 + (size_t)r * BPER, nullptr, out2, 0);
  }
  k_convA<<<convBlocks, 256, 0, stream>>>(cur1, nullptr, Ah);
  k_gemm_tile<<<gGrid, 32, 0, stream>>>(Ah, Brs2, bm2, out2, 1);
  k_gate<<<(NN + 255) / 256, 256, 0, stream>>>(out2, past2, sv2, snap, cur2);

  // ---- relation representation + decoder ----
  k_avg<<<dim3(KD, RR), 256, 0, stream>>>(cur2, ei, avg);
  k_gru<<<RR, 256, 0, stream>>>(avg, rel, pastR, gWih, gWhh, gbih, gbhh, lW, lb, snap, curR);
  k_rterm<<<RR, 64, 0, stream>>>(curR, crW, crb, pwb, rterm);
  k_decode<<<dim3((ELL + 127) / 128, RR), 128, 0, stream>>>(cur2, eli, chW, chb, pwb,
                                                            rterm, out);
}
